// Mamba2LMHeadModel_26757646254671
// MI455X (gfx1250) — compile-verified, matrix operand order checked by reading, not run on hardware
//
#include <hip/hip_runtime.h>

// ---------------------------------------------------------------------------
// Mamba-2 SSD chunked scan + learned lower-tri mask, MI455X (gfx1250, wave32)
//
// b=2 T=1024 h=32 p=64 n=128 l=64 k=16.  Memory-bound on L_mask (256MB) =>
// stream L_mask from HBM; all matmuls via v_wmma_f32_16x16x32_f16 (f32 accum).
// All LDS tiles are stored in WMMA *fragment-major* order so each lane loads
// its v16h operand with 2x ds_load_b128 instead of 16x ds_load_u16.
// ---------------------------------------------------------------------------

typedef __attribute__((ext_vector_type(16))) _Float16 v16h;
typedef __attribute__((ext_vector_type(4)))  _Float16 v4h;
typedef __attribute__((ext_vector_type(8)))  float    v8f;

constexpr int B_ = 2;
constexpr int T_ = 1024;
constexpr int H_ = 32;
constexpr int P_ = 64;
constexpr int N_ = 128;
constexpr int L_ = 64;     // chunk size
constexpr int K_ = 16;     // number of chunks

// ISA 7.12.2 16-bit A-matrix K pattern: lane = g*16+m,
//   k(e,g) = (e<8) ? g*8+e : 16+g*8+(e-8)
// Inverse (for staging writers): given kk in 0..31:
//   g = (kk>>3)&1 ;  e = (kk&7) + ((kk>>4)<<3)
// Fragment-major LDS layout: frag[tile][ks][lane][e], 16 e contiguous.
__device__ __forceinline__ int frag_off(int rr, int kk, int nks) {
  int tile = rr >> 4, m = rr & 15;
  int ks = kk >> 5, kl = kk & 31;
  int g = (kl >> 3) & 1;
  int e = (kl & 7) + ((kl >> 4) << 3);
  return ((tile * nks + ks) * 32 + (g * 16 + m)) * 16 + e;
}

__device__ __forceinline__ v16h frag_ld(const _Float16* fbase, int tile,
                                        int nks, int ks, int lane) {
  return *(const v16h*)(fbase + ((tile * nks + ks) * 32 + lane) * 16);
}

// ---------------------------------------------------------------------------
// Phase 0: inclusive cumsum of A within each chunk.  acs[b][h][c][l]
// ---------------------------------------------------------------------------
__global__ void k_acs(const float* __restrict__ A, float* __restrict__ acs) {
  int c = blockIdx.x % K_;
  int h = (blockIdx.x / K_) % H_;
  int b = blockIdx.x / (K_ * H_);
  int t = threadIdx.x;  // 0..63
  float s = 0.f;
  for (int j = 0; j <= t; ++j)
    s += A[((size_t)b * T_ + c * L_ + j) * H_ + h];
  acs[((size_t)(b * H_ + h) * K_ + c) * L_ + t] = s;
}

// ---------------------------------------------------------------------------
// Phase 1: states[p][n] = sum_l exp(Acs[63]-Acs[l]) * x[l][p] * B[l][n]
// One WG per (b,c,h); 512 threads = 16 waves; 64x128 D = 4x8 tiles, 2/wave.
// ---------------------------------------------------------------------------
__global__ void __launch_bounds__(512)
k_states(const float* __restrict__ x, const float* __restrict__ Bm,
         const float* __restrict__ acs, float* __restrict__ states) {
  __shared__ __align__(32) _Float16 ahF[4 * 2 * 32 * 16];  // A-role: rr=p, kk=l
  __shared__ __align__(32) _Float16 bhF[8 * 2 * 32 * 16];  // B-role: rr=n, kk=l
  __shared__ float sc[L_];

  int c = blockIdx.x % K_;
  int h = (blockIdx.x / K_) % H_;
  int b = blockIdx.x / (K_ * H_);
  int tid = threadIdx.x;

  size_t ab = ((size_t)(b * H_ + h) * K_ + c) * L_;
  if (tid < L_) sc[tid] = acs[ab + tid];
  __syncthreads();
  float alast = sc[L_ - 1];

  // stage weighted x (transposed role): global float4 over p, scatter f16
  for (int i = tid; i < (L_ * P_) / 4; i += 512) {
    int e0 = i * 4;
    int l = e0 >> 6, p = e0 & 63;
    float w = __expf(alast - sc[l]);
    const float4 v =
        *(const float4*)&x[(((size_t)b * T_ + c * L_ + l) * H_ + h) * P_ + p];
    ahF[frag_off(p + 0, l, 2)] = (_Float16)(v.x * w);
    ahF[frag_off(p + 1, l, 2)] = (_Float16)(v.y * w);
    ahF[frag_off(p + 2, l, 2)] = (_Float16)(v.z * w);
    ahF[frag_off(p + 3, l, 2)] = (_Float16)(v.w * w);
  }
  for (int i = tid; i < (L_ * N_) / 4; i += 512) {
    int e0 = i * 4;
    int l = e0 >> 7, n = e0 & 127;
    const float4 v =
        *(const float4*)&Bm[(((size_t)b * T_ + c * L_ + l) * H_ + h) * N_ + n];
    bhF[frag_off(n + 0, l, 2)] = (_Float16)v.x;
    bhF[frag_off(n + 1, l, 2)] = (_Float16)v.y;
    bhF[frag_off(n + 2, l, 2)] = (_Float16)v.z;
    bhF[frag_off(n + 3, l, 2)] = (_Float16)v.w;
  }
  __syncthreads();

  int wv = tid >> 5, m = tid & 15, lane = tid & 31;
  int pi = wv >> 2;
  size_t base = ((size_t)(b * H_ + h) * K_ + c) * P_ * N_;
#pragma unroll
  for (int jj = 0; jj < 2; ++jj) {
    int nj = (wv & 3) * 2 + jj;
    v8f acc = {};
#pragma unroll
    for (int ks = 0; ks < 2; ++ks) {  // K = 64
      v16h a = frag_ld(ahF, pi, 2, ks, lane);
      v16h bb = frag_ld(bhF, nj, 2, ks, lane);
      acc = __builtin_amdgcn_wmma_f32_16x16x32_f16(false, a, false, bb,
                                                   (short)0, acc, false, false);
    }
    int g = (tid >> 4) & 1;
#pragma unroll
    for (int r = 0; r < 8; ++r) {
      int p = pi * 16 + r + 8 * g;  // C/D layout: row = vgpr + 8*lanegroup
      int n = nj * 16 + m;          // col = lane&15
      states[base + (size_t)p * N_ + n] = acc[r];
    }
  }
}

// ---------------------------------------------------------------------------
// Phase 2: stable sequential chunk recurrence per (b,h):
//   R_z = exp(a_z)*R_{z-1} + chunkstate[z-1]
// new_states[z] (z=1..15) written IN-PLACE into slot z-1; phase 3 reads
// states_in[c] from slot c-1 (c=0 -> zero, skipped).  z=16 -> final_state.
// ---------------------------------------------------------------------------
__global__ void __launch_bounds__(256)
k_chain(const float* __restrict__ acs, float* __restrict__ states,
        float* __restrict__ fs) {
  int h = blockIdx.x % H_;
  int b = blockIdx.x / H_;
  int tid = threadIdx.x;
  size_t sb = (size_t)(b * H_ + h) * K_ * P_ * N_;
  float4 R[8];
#pragma unroll
  for (int j = 0; j < 8; ++j) R[j] = make_float4(0.f, 0.f, 0.f, 0.f);
  for (int z = 1; z <= K_; ++z) {
    float az = acs[((size_t)(b * H_ + h) * K_ + (z - 1)) * L_ + (L_ - 1)];
    float w = __expf(az);
    float4* slot = (float4*)(states + sb + (size_t)(z - 1) * P_ * N_);
#pragma unroll
    for (int j = 0; j < 8; ++j) {  // 256 thr * 8 * 4 = 8192 = P*N
      float4 cs = slot[tid + 256 * j];
      R[j].x = w * R[j].x + cs.x;
      R[j].y = w * R[j].y + cs.y;
      R[j].z = w * R[j].z + cs.z;
      R[j].w = w * R[j].w + cs.w;
      if (z < K_) slot[tid + 256 * j] = R[j];
    }
  }
  float4* ob = (float4*)(fs + (size_t)(b * H_ + h) * P_ * N_);
#pragma unroll
  for (int j = 0; j < 8; ++j) ob[tid + 256 * j] = R[j];
}

// ---------------------------------------------------------------------------
// Phase 3: per (b,h,c) WG computes the 64x64 output chunk:
//   for s=0..c: S = C_c B_s^T (K=128); P = mask .* S; Y += P x_s (K=64)
//     mask = exp(segsum) tril + msf*Lmask (s==c) | msf*Lmask (s<c)
//   Y += exp(Acs[l]) * (C_c states_in[c]^T)   (K=128)
// 512 threads = 16 waves, one 16x16 Y tile each.
// ---------------------------------------------------------------------------
__global__ void __launch_bounds__(512)
k_attn(const float* __restrict__ x, const float* __restrict__ Bm,
       const float* __restrict__ Cm, const float* __restrict__ Lmask,
       const float* __restrict__ msfp, const float* __restrict__ acs,
       const float* __restrict__ states, float* __restrict__ Y) {
  __shared__ __align__(32) _Float16 chF[4 * 4 * 32 * 16];  // C_c  (rr=l, kk=n)
  __shared__ __align__(32) _Float16 bhF[4 * 4 * 32 * 16];  // B_s / states_in
  __shared__ __align__(32) _Float16 xhF[4 * 2 * 32 * 16];  // x_s  (rr=p, kk=l)
  __shared__ __align__(32) _Float16 phF[4 * 2 * 32 * 16];  // P    (rr=l, kk=j)
  __shared__ float sc[L_];

  int c = blockIdx.x % K_;
  int h = (blockIdx.x / K_) % H_;
  int b = blockIdx.x / (K_ * H_);
  int tid = threadIdx.x;
  float msf = msfp[0];

  size_t ab = ((size_t)(b * H_ + h) * K_ + c) * L_;
  if (tid < L_) sc[tid] = acs[ab + tid];
  for (int i = tid; i < (L_ * N_) / 4; i += 512) {
    int e0 = i * 4;
    int l = e0 >> 7, n = e0 & 127;
    const float4 v =
        *(const float4*)&Cm[(((size_t)b * T_ + c * L_ + l) * H_ + h) * N_ + n];
    v4h pk;
    pk[0] = (_Float16)v.x; pk[1] = (_Float16)v.y;
    pk[2] = (_Float16)v.z; pk[3] = (_Float16)v.w;
    *(v4h*)(chF + frag_off(l, n, 4)) = pk;  // 4 consecutive kk share a slot
  }

  int wv = tid >> 5, m = tid & 15, g = (tid >> 4) & 1, lane = tid & 31;
  int ti = wv >> 2, tj = wv & 3;
  v8f yacc = {};
  size_t lmb = (size_t)(b * H_ + h) * T_ * T_;

  for (int s = 0; s <= c; ++s) {
    __syncthreads();  // prior iteration's readers of bhF/xhF/phF are done
    for (int i = tid; i < (L_ * N_) / 4; i += 512) {
      int e0 = i * 4;
      int l = e0 >> 7, n = e0 & 127;
      const float4 v =
          *(const float4*)&Bm[(((size_t)b * T_ + s * L_ + l) * H_ + h) * N_ + n];
      v4h pk;
      pk[0] = (_Float16)v.x; pk[1] = (_Float16)v.y;
      pk[2] = (_Float16)v.z; pk[3] = (_Float16)v.w;
      *(v4h*)(bhF + frag_off(l, n, 4)) = pk;
    }
    for (int i = tid; i < (L_ * P_) / 4; i += 512) {
      int e0 = i * 4;
      int l = e0 >> 6, p = e0 & 63;
      const float4 v =
          *(const float4*)&x[(((size_t)b * T_ + s * L_ + l) * H_ + h) * P_ + p];
      xhF[frag_off(p + 0, l, 2)] = (_Float16)v.x;  // B-col role: rr=p
      xhF[frag_off(p + 1, l, 2)] = (_Float16)v.y;
      xhF[frag_off(p + 2, l, 2)] = (_Float16)v.z;
      xhF[frag_off(p + 3, l, 2)] = (_Float16)v.w;
    }
    if (s < c)  // warm next B chunk (global_prefetch_b8)
      __builtin_prefetch(&Bm[(((size_t)b * T_ + (s + 1) * L_) * H_ + h) * N_], 0, 1);
    __syncthreads();

    // scores S[l][j] = sum_n C[l][n] * B[j][n]
    v8f sacc = {};
#pragma unroll
    for (int ks = 0; ks < 4; ++ks) {  // K = 128
      v16h a = frag_ld(chF, ti, 4, ks, lane);
      v16h bb = frag_ld(bhF, tj, 4, ks, lane);
      sacc = __builtin_amdgcn_wmma_f32_16x16x32_f16(false, a, false, bb,
                                                    (short)0, sacc, false, false);
    }
    // mask (f32), streaming L_mask straight from HBM (coalesced 16-lane rows)
#pragma unroll
    for (int r = 0; r < 8; ++r) {
      int li = ti * 16 + r + 8 * g;
      int jj = tj * 16 + m;
      float lm = Lmask[lmb + (size_t)(c * L_ + li) * T_ + s * L_ + jj];
      float mask = (s == c)
          ? (((li >= jj) ? __expf(sc[li] - sc[jj]) : 0.f) + msf * lm)
          : (msf * lm);
      phF[frag_off(li, jj, 2)] = (_Float16)(sacc[r] * mask);
    }
    __syncthreads();

    // Y += P * x_s   (K = 64)
#pragma unroll
    for (int ks = 0; ks < 2; ++ks) {
      v16h a = frag_ld(phF, ti, 2, ks, lane);
      v16h bb = frag_ld(xhF, tj, 2, ks, lane);
      yacc = __builtin_amdgcn_wmma_f32_16x16x32_f16(false, a, false, bb,
                                                    (short)0, yacc, false, false);
    }
  }

  // Y_off: states_in[c] is in slot c-1 (c==0 -> zero contribution)
  if (c > 0) {
    __syncthreads();
    const float* st = states + ((size_t)(b * H_ + h) * K_ + (c - 1)) * P_ * N_;
    for (int i = tid; i < (P_ * N_) / 4; i += 512) {
      int e0 = i * 4;
      int p = e0 >> 7, n = e0 & 127;
      const float4 v = *(const float4*)&st[e0];
      v4h pk;
      pk[0] = (_Float16)v.x; pk[1] = (_Float16)v.y;
      pk[2] = (_Float16)v.z; pk[3] = (_Float16)v.w;
      *(v4h*)(bhF + frag_off(p, n, 4)) = pk;  // B-role: rr=p, kk=n
    }
    __syncthreads();
    v8f tacc = {};
#pragma unroll
    for (int ks = 0; ks < 4; ++ks) {  // K = 128
      v16h a = frag_ld(chF, ti, 4, ks, lane);
      v16h bb = frag_ld(bhF, tj, 4, ks, lane);
      tacc = __builtin_amdgcn_wmma_f32_16x16x32_f16(false, a, false, bb,
                                                    (short)0, tacc, false, false);
    }
#pragma unroll
    for (int r = 0; r < 8; ++r) {
      int li = ti * 16 + r + 8 * g;
      yacc[r] += __expf(sc[li]) * tacc[r];
    }
  }

  // write Y[b, c*L+l, h, p]
#pragma unroll
  for (int r = 0; r < 8; ++r) {
    int li = ti * 16 + r + 8 * g;
    int p = tj * 16 + m;
    Y[(((size_t)b * T_ + c * L_ + li) * H_ + h) * P_ + p] = yacc[r];
  }
}

// ---------------------------------------------------------------------------
extern "C" void kernel_launch(void* const* d_in, const int* in_sizes, int n_in,
                              void* d_out, int out_size, void* d_ws,
                              size_t ws_size, hipStream_t stream) {
  (void)in_sizes; (void)n_in; (void)out_size; (void)ws_size;
  const float* x   = (const float*)d_in[0];
  const float* A   = (const float*)d_in[1];
  const float* Bm  = (const float*)d_in[2];
  const float* Cm  = (const float*)d_in[3];
  const float* Lm  = (const float*)d_in[4];
  const float* msf = (const float*)d_in[5];
  // d_in[6] = chunk_size (64, hardcoded)

  float* Y  = (float*)d_out;
  float* FS = Y + (size_t)B_ * T_ * H_ * P_;  // final_state after Y

  // workspace: Acs (256KB) + states (32MB)
  float* acs    = (float*)d_ws;
  float* states = acs + (size_t)B_ * H_ * K_ * L_;

  k_acs   <<<B_ * H_ * K_,  64, 0, stream>>>(A, acs);
  k_states<<<B_ * H_ * K_, 512, 0, stream>>>(x, Bm, acs, states);
  k_chain <<<B_ * H_,      256, 0, stream>>>(acs, states, FS);
  k_attn  <<<B_ * H_ * K_, 512, 0, stream>>>(x, Bm, Cm, Lm, msf, acs, states, Y);
}
